// ASSM_52080773431761
// MI455X (gfx1250) — compile-verified, matrix operand order checked by reading, not run on hardware
//
#include <hip/hip_runtime.h>
#include <hip/hip_bf16.h>
#include <stdint.h>

#define Bb     2
#define Nn     9216          // 4*48*48
#define DIMC   96
#define HID    192
#define DST    16
#define DTR    12
#define NTOK   2048
#define IRANK  128
#define DD_    4
#define HH_    48
#define WW_    48
#define SCT    64            // scan chunk length (steps staged per async batch)

typedef _Float16 v16h __attribute__((ext_vector_type(16)));
typedef float    v8f  __attribute__((ext_vector_type(8)));

// ---- CDNA5 WMMA 16x16x32 f16 fragment index maps (ISA 7.12.2, 16-bit A/B) ----
__device__ __forceinline__ int kmap(int e, int hs) {
  return (e < 8 ? e : e + 8) + 8 * hs;
}
__device__ __forceinline__ void amap(int kk, int m, int& lane, int& e) {
  int hs, ee;
  if      (kk < 8)  { hs = 0; ee = kk;      }
  else if (kk < 16) { hs = 1; ee = kk - 8;  }
  else if (kk < 24) { hs = 0; ee = kk - 8;  }
  else              { hs = 1; ee = kk - 16; }
  lane = hs * 16 + m; e = ee;
}
__device__ __forceinline__ v8f wmma32(v16h a, v16h b, v8f c) {
  return __builtin_amdgcn_wmma_f32_16x16x32_f16(false, a, false, b, (short)0, c,
                                                false, false);
}

// ---- pack a row-major f32 weight (O x K) into B-fragment layout, zero-padded ----
__global__ void k_pack_bfrag(const float* __restrict__ W, _Float16* __restrict__ frag,
                             int O, int K, int kslabs, int total) {
  int tid = blockIdx.x * blockDim.x + threadIdx.x;
  if (tid >= total) return;
  int e = tid & 15, lane = (tid >> 4) & 31, fidx = tid >> 9;
  int ot = fidx / kslabs, ks = fidx % kslabs;
  int n = lane & 15, hs = lane >> 4;
  int o  = ot * 16 + n;
  int kk = ks * 32 + kmap(e, hs);
  float v = (o < O && kk < K) ? W[(size_t)o * K + kk] : 0.0f;
  frag[tid] = (_Float16)v;
}

// ---- full_emb = embB @ token_weight : (2048,16) ----
__global__ void k_full_emb(const float* __restrict__ embB, const float* __restrict__ tw,
                           float* __restrict__ fe) {
  int gid = blockIdx.x * blockDim.x + threadIdx.x;
  if (gid >= NTOK * DST) return;
  int t = gid >> 4, n = gid & 15;
  float acc = 0.f;
  for (int r = 0; r < IRANK; ++r) acc += embB[t * IRANK + r] * tw[r * DST + n];
  fe[gid] = acc;
}

// ---- routing: gelu(x@W1^T+b1) -> WMMA logits over 2048 classes -> gumbel argmax ----
__global__ void k_route(const float* __restrict__ x, const float* __restrict__ w1,
                        const float* __restrict__ b1, const float* __restrict__ b2,
                        const _Float16* __restrict__ fragr2,
                        unsigned long long* __restrict__ argbuf) {
  __shared__ __align__(32) _Float16 h1f[512];
  int b = blockIdx.y, tile0 = blockIdx.x * 16, tid = threadIdx.x;
  for (int i = tid; i < 512; i += 256) {
    int m = i >> 5, o = i & 31, l = tile0 + m;
    const float* xr = x + ((size_t)b * Nn + l) * DIMC;
    const float* wr = w1 + o * DIMC;
    float acc = b1[o];
    for (int k = 0; k < DIMC; ++k) acc += xr[k] * wr[k];
    float t = 0.7978845608028654f * (acc + 0.044715f * acc * acc * acc);
    float g = 0.5f * acc * (1.0f + tanhf(t));   // tanh-approx gelu (JAX default)
    int ln, e; amap(o, m, ln, e);
    h1f[ln * 16 + e] = (_Float16)g;
  }
  __syncthreads();
  int wave = tid >> 5, lane = tid & 31;
  v16h a = *reinterpret_cast<const v16h*>(h1f + lane * 16);
  for (int it = 0; it < 16; ++it) {
    int ctile = it * 8 + wave;
    v16h bf = *reinterpret_cast<const v16h*>(fragr2 + ((size_t)ctile * 32 + lane) * 16);
    v8f c = {};
    c = wmma32(a, bf, c);
    int cls = ctile * 16 + (lane & 15);
    float bias = b2[cls];
    for (int v = 0; v < 8; ++v) {
      int m = v + 8 * (lane >> 4), l = tile0 + m;
      unsigned s = (unsigned)((b * Nn + l) * 2048 + cls);
      s ^= s >> 16; s *= 0x7feb352du; s ^= s >> 15; s *= 0x846ca68bu; s ^= s >> 16;
      float u   = (float)(s >> 8) * (1.0f / 16777216.0f) + 1e-20f;
      float gum = -logf(-logf(u));
      float val = c[v] + bias + gum;
      unsigned fb = __float_as_uint(val);
      fb = (fb & 0x80000000u) ? ~fb : (fb | 0x80000000u);
      unsigned long long packed = ((unsigned long long)fb << 32) | (unsigned)cls;
      atomicMax(&argbuf[(size_t)b * Nn + l], packed);
    }
  }
}

__global__ void k_extract(const unsigned long long* __restrict__ argbuf,
                          int* __restrict__ idx, int total) {
  int i = blockIdx.x * blockDim.x + threadIdx.x;
  if (i < total) idx[i] = (int)(argbuf[i] & 0xFFFFFFFFull);
}

__global__ void k_hist(const int* __restrict__ idx, int* __restrict__ hist) {
  int i = blockIdx.x * blockDim.x + threadIdx.x;
  if (i < Bb * Nn) { int b = i / Nn; atomicAdd(&hist[b * NTOK + idx[i]], 1); }
}

__global__ void k_offsets(const int* __restrict__ hist, int* __restrict__ offs) {
  int b = blockIdx.x;
  if (threadIdx.x == 0) {
    int acc = 0;
    for (int t = 0; t < NTOK; ++t) { offs[b * NTOK + t] = acc; acc += hist[b * NTOK + t]; }
  }
}

// stable counting-sort placement: one thread per bin, keys staged through LDS
__global__ void k_place(const int* __restrict__ idx, const int* __restrict__ offs,
                        int* __restrict__ sidx, int* __restrict__ keys) {
  __shared__ int chunk[256];
  int b = blockIdx.x;
  int bin = blockIdx.y * 256 + threadIdx.x;
  int off = offs[b * NTOK + bin];
  const int* ip = idx  + (size_t)b * Nn;
  int* sp = sidx + (size_t)b * Nn;
  int* kp = keys + (size_t)b * Nn;
  for (int l0 = 0; l0 < Nn; l0 += 256) {
    chunk[threadIdx.x] = ip[l0 + threadIdx.x];
    __syncthreads();
    for (int t = 0; t < 256; ++t)
      if (chunk[t] == bin) { sp[off] = l0 + t; kp[off] = bin; ++off; }
    __syncthreads();
  }
}

// ---- inproj: (B,N,96) x (192,96)^T via WMMA, K=96 = 3 slabs ----
__global__ void k_inproj(const float* __restrict__ x, const float* __restrict__ ipb,
                         const _Float16* __restrict__ frag, float* __restrict__ xin) {
  __shared__ __align__(32) _Float16 af[512];
  int b = blockIdx.y, tile0 = blockIdx.x * 16, tid = threadIdx.x;
  int wave = tid >> 5, lane = tid & 31;
  v8f c = {};
  for (int ks = 0; ks < 3; ++ks) {
    for (int i = tid; i < 512; i += 384) {
      int m = i >> 5, kk = i & 31;
      float v = x[((size_t)b * Nn + tile0 + m) * DIMC + ks * 32 + kk];
      int ln, e; amap(kk, m, ln, e);
      af[ln * 16 + e] = (_Float16)v;
    }
    __syncthreads();
    v16h a  = *reinterpret_cast<const v16h*>(af + lane * 16);
    v16h bf = *reinterpret_cast<const v16h*>(frag + ((size_t)(wave * 3 + ks) * 32 + lane) * 16);
    c = wmma32(a, bf, c);
    __syncthreads();
  }
  int ch = wave * 16 + (lane & 15);
  for (int v = 0; v < 8; ++v) {
    int m = v + 8 * (lane >> 4);
    xin[((size_t)b * Nn + tile0 + m) * HID + ch] = c[v] + ipb[ch];
  }
}

// ---- depthwise 3x3x3 CPE + x*sigmoid gate ----
__global__ void k_cpe(const float* __restrict__ xin, const float* __restrict__ cw,
                      const float* __restrict__ cb, float* __restrict__ xg) {
  size_t i = (size_t)blockIdx.x * blockDim.x + threadIdx.x;
  size_t total = (size_t)Bb * Nn * HID;
  if (i >= total) return;
  int c = (int)(i % HID);
  size_t bl = i / HID;
  int l = (int)(bl % Nn), b = (int)(bl / Nn);
  int w = l % WW_, h = (l / WW_) % HH_, d = l / (WW_ * HH_);
  float acc = cb[c];
  for (int kd = 0; kd < 3; ++kd) {
    int dd = d + kd - 1; if (dd < 0 || dd >= DD_) continue;
    for (int kh = 0; kh < 3; ++kh) {
      int hh = h + kh - 1; if (hh < 0 || hh >= HH_) continue;
      for (int kw = 0; kw < 3; ++kw) {
        int ww = w + kw - 1; if (ww < 0 || ww >= WW_) continue;
        int ln = (dd * HH_ + hh) * WW_ + ww;
        acc += xin[((size_t)b * Nn + ln) * HID + c] * cw[c * 27 + (kd * 3 + kh) * 3 + kw];
      }
    }
  }
  float xv = xin[i];
  xg[i] = xv / (1.0f + __expf(-acc));
}

// ---- gather + x_proj (K=192, 44->48 padded) + dt (K=12 padded to 32) + softplus ----
__global__ void k_xproj(const float* __restrict__ xg, const int* __restrict__ sidx_g,
                        const int* __restrict__ keys_g,
                        const _Float16* __restrict__ fragxp,
                        const _Float16* __restrict__ fragdt,
                        const float* __restrict__ fe, const float* __restrict__ dt_b,
                        float* __restrict__ xs_s, float* __restrict__ delta_s,
                        float* __restrict__ Bs_s, float* __restrict__ Cs_s) {
  __shared__ __align__(32) _Float16 af[512];
  __shared__ __align__(32) _Float16 af2[512];
  __shared__ float xdbl[16][48];
  __shared__ int sidx[16];
  __shared__ int keys[16];
  int b = blockIdx.y, j0 = blockIdx.x * 16, tid = threadIdx.x;
  int wave = tid >> 5, lane = tid & 31;
  if (tid < 16) {
    sidx[tid] = sidx_g[(size_t)b * Nn + j0 + tid];
    keys[tid] = keys_g[(size_t)b * Nn + j0 + tid];
  }
  __syncthreads();
  v8f c1 = {};
  for (int ks = 0; ks < 6; ++ks) {
    for (int i = tid; i < 512; i += 384) {
      int m = i >> 5, kk = i & 31, ch = ks * 32 + kk;
      float v = xg[((size_t)b * Nn + sidx[m]) * HID + ch];
      xs_s[((size_t)b * Nn + j0 + m) * HID + ch] = v;
      int ln, e; amap(kk, m, ln, e);
      af[ln * 16 + e] = (_Float16)v;
    }
    __syncthreads();
    if (wave < 3) {
      v16h a  = *reinterpret_cast<const v16h*>(af + lane * 16);
      v16h bf = *reinterpret_cast<const v16h*>(fragxp + ((size_t)(wave * 6 + ks) * 32 + lane) * 16);
      c1 = wmma32(a, bf, c1);
    }
    __syncthreads();
  }
  if (wave < 3) {
    int ch = wave * 16 + (lane & 15);
    for (int v = 0; v < 8; ++v) { int m = v + 8 * (lane >> 4); xdbl[m][ch] = c1[v]; }
  }
  __syncthreads();
  for (int i = tid; i < 256; i += 384) {
    int m = i >> 4, n = i & 15;
    Bs_s[((size_t)b * Nn + j0 + m) * DST + n] = xdbl[m][DTR + n];
    Cs_s[((size_t)b * Nn + j0 + m) * DST + n] = xdbl[m][DTR + DST + n] + fe[keys[m] * DST + n];
  }
  for (int i = tid; i < 512; i += 384) {
    int m = i >> 5, kk = i & 31;
    float v = (kk < DTR) ? xdbl[m][kk] : 0.0f;
    int ln, e; amap(kk, m, ln, e);
    af2[ln * 16 + e] = (_Float16)v;
  }
  __syncthreads();
  v16h a2 = *reinterpret_cast<const v16h*>(af2 + lane * 16);
  v16h bf = *reinterpret_cast<const v16h*>(fragdt + ((size_t)wave * 32 + lane) * 16);
  v8f c2 = {};
  c2 = wmma32(a2, bf, c2);
  int ch = wave * 16 + (lane & 15);
  float db = dt_b[ch];
  for (int v = 0; v < 8; ++v) {
    int m = v + 8 * (lane >> 4);
    float dv = c2[v] + db;
    float delta = dv > 20.f ? dv : log1pf(__expf(dv));
    delta_s[((size_t)b * Nn + j0 + m) * HID + ch] = delta;
  }
}

// ---- selective scan: lane = (d,n); double-buffered async-to-LDS chunk staging ----
// Per chunk: 4 segments (delta, xs, Bs, Cs) x SCT steps x 16 lanes = 4096 floats.
// global_load_async_to_lds_b32 (ASYNCcnt) prefetches chunk c+1 while chunk c is
// consumed from LDS; async loads complete in order, so waiting ASYNCcnt<=16
// (the 16 just-issued next-chunk loads) guarantees the current chunk landed.
__global__ void k_scan(const float* __restrict__ A_logs, const float* __restrict__ Ds,
                       const float* __restrict__ xs_s, const float* __restrict__ delta_s,
                       const float* __restrict__ Bs_s, const float* __restrict__ Cs_s,
                       float* __restrict__ y_s) {
  __shared__ float buf[2][4][SCT][16];
  int b = blockIdx.y;
  int dbase = blockIdx.x * 16;
  int tid = threadIdx.x;
  int dloc = tid >> 4, n = tid & 15;
  int d = dbase + dloc;
  float Adn = -__expf(A_logs[d * DST + n]);
  float Dd  = Ds[d];
  float h = 0.f;
  const size_t bofH = (size_t)b * Nn * HID;
  const size_t bofS = (size_t)b * Nn * DST;
  float* yp = y_s + bofH + d;

  auto issue = [&](int l0, int sel) {
    for (int q = 0; q < 16; ++q) {
      int idx = q * 256 + tid;             // 0..4095
      int seg = idx >> 10;                 // 0..3
      int t   = (idx >> 4) & (SCT - 1);
      int j   = idx & 15;
      const float* g;
      if      (seg == 0) g = delta_s + bofH + (size_t)(l0 + t) * HID + dbase + j;
      else if (seg == 1) g = xs_s    + bofH + (size_t)(l0 + t) * HID + dbase + j;
      else if (seg == 2) g = Bs_s    + bofS + (size_t)(l0 + t) * DST + j;
      else               g = Cs_s    + bofS + (size_t)(l0 + t) * DST + j;
      unsigned loff = (unsigned)(uintptr_t)&buf[sel][seg][t][j];  // LDS byte offset
      asm volatile("global_load_async_to_lds_b32 %0, %1, off"
                   :: "v"(loff), "v"(g) : "memory");
    }
  };

  issue(0, 0);
  const int nch = Nn / SCT;
  for (int c = 0; c < nch; ++c) {
    int sel = c & 1;
    if (c + 1 < nch) {
      issue((c + 1) * SCT, sel ^ 1);
      asm volatile("s_wait_asynccnt 0x10" ::: "memory");  // current chunk complete
    } else {
      asm volatile("s_wait_asynccnt 0x0" ::: "memory");
    }
    __syncthreads();
    int l0 = c * SCT;
    for (int t = 0; t < SCT; ++t) {
      float delta = buf[sel][0][t][dloc];
      float xv    = buf[sel][1][t][dloc];
      float Bv    = buf[sel][2][t][n];
      float Cv    = buf[sel][3][t][n];
      h = __expf(delta * Adn) * h + (delta * xv) * Bv;
      float p = h * Cv;
      p += __shfl_xor(p, 1, 16);
      p += __shfl_xor(p, 2, 16);
      p += __shfl_xor(p, 4, 16);
      p += __shfl_xor(p, 8, 16);
      if (n == 0) yp[(size_t)(l0 + t) * HID] = p + xv * Dd;
    }
    __syncthreads();   // all lanes done reading buf[sel] before it is refilled
  }
}

// ---- LayerNorm + out-proj (WMMA, K=192) + scatter by sort_idx ----
__global__ void k_out(const float* __restrict__ y_s, const float* __restrict__ ng,
                      const float* __restrict__ nb, const _Float16* __restrict__ fragout,
                      const float* __restrict__ ob, const int* __restrict__ sidx_g,
                      float* __restrict__ out) {
  __shared__ float yr[16][HID];
  __shared__ __align__(32) _Float16 af[6][512];
  __shared__ float mv[16][2];
  __shared__ int sidx[16];
  int b = blockIdx.y, j0 = blockIdx.x * 16, tid = threadIdx.x;
  if (tid < 16) sidx[tid] = sidx_g[(size_t)b * Nn + j0 + tid];
  for (int i = tid; i < 16 * HID; i += 256) {
    int m = i / HID, ch = i % HID;
    yr[m][ch] = y_s[((size_t)b * Nn + j0 + m) * HID + ch];
  }
  __syncthreads();
  {
    int m = tid >> 4, g = tid & 15;
    float s1 = 0.f, s2 = 0.f;
    for (int ch = g; ch < HID; ch += 16) { float v = yr[m][ch]; s1 += v; s2 += v * v; }
    s1 += __shfl_xor(s1, 1, 16); s2 += __shfl_xor(s2, 1, 16);
    s1 += __shfl_xor(s1, 2, 16); s2 += __shfl_xor(s2, 2, 16);
    s1 += __shfl_xor(s1, 4, 16); s2 += __shfl_xor(s2, 4, 16);
    s1 += __shfl_xor(s1, 8, 16); s2 += __shfl_xor(s2, 8, 16);
    if (g == 0) {
      float mean = s1 / HID;
      float var  = s2 / HID - mean * mean;
      mv[m][0] = mean; mv[m][1] = rsqrtf(var + 1e-5f);
    }
  }
  __syncthreads();
  for (int i = tid; i < 16 * HID; i += 256) {
    int m = i / HID, ch = i % HID;
    float v = (yr[m][ch] - mv[m][0]) * mv[m][1] * ng[ch] + nb[ch];
    int ln, e; amap(ch & 31, m, ln, e);
    af[ch >> 5][ln * 16 + e] = (_Float16)v;
  }
  __syncthreads();
  int wave = tid >> 5, lane = tid & 31;
  if (wave < 6) {
    v8f c = {};
    for (int ks = 0; ks < 6; ++ks) {
      v16h a  = *reinterpret_cast<const v16h*>(&af[ks][lane * 16]);
      v16h bf = *reinterpret_cast<const v16h*>(fragout + ((size_t)(wave * 6 + ks) * 32 + lane) * 16);
      c = wmma32(a, bf, c);
    }
    int ch = wave * 16 + (lane & 15);
    float bias = ob[ch];
    for (int v = 0; v < 8; ++v) {
      int m = v + 8 * (lane >> 4);
      out[((size_t)b * Nn + sidx[m]) * DIMC + ch] = c[v] + bias;
    }
  }
}

extern "C" void kernel_launch(void* const* d_in, const int* in_sizes, int n_in,
                              void* d_out, int out_size, void* d_ws, size_t ws_size,
                              hipStream_t stream) {
  (void)in_sizes; (void)n_in; (void)out_size; (void)ws_size;
  const float* x    = (const float*)d_in[0];
  const float* tw   = (const float*)d_in[1];
  const float* embB = (const float*)d_in[2];
  const float* rw1  = (const float*)d_in[3];
  const float* rb1  = (const float*)d_in[4];
  const float* rw2  = (const float*)d_in[5];
  const float* rb2  = (const float*)d_in[6];
  const float* ipw  = (const float*)d_in[7];
  const float* ipb  = (const float*)d_in[8];
  const float* cpw  = (const float*)d_in[9];
  const float* cpb  = (const float*)d_in[10];
  const float* xpw  = (const float*)d_in[11];
  const float* dtw  = (const float*)d_in[12];
  const float* dtb  = (const float*)d_in[13];
  const float* Alog = (const float*)d_in[14];
  const float* Ds   = (const float*)d_in[15];
  const float* ng   = (const float*)d_in[16];
  const float* nb   = (const float*)d_in[17];
  const float* ow   = (const float*)d_in[18];
  const float* ob   = (const float*)d_in[19];
  float* out = (float*)d_out;
  char*  ws  = (char*)d_ws;

  float*              fe       = (float*)(ws + 0);
  _Float16*           frag_r2  = (_Float16*)(ws + 131072);
  _Float16*           frag_in  = (_Float16*)(ws + 262144);
  _Float16*           frag_xp  = (_Float16*)(ws + 299008);
  _Float16*           frag_dt  = (_Float16*)(ws + 317440);
  _Float16*           frag_out = (_Float16*)(ws + 329728);
  unsigned long long* argbuf   = (unsigned long long*)(ws + 366592);
  int*                idx      = (int*)(ws + 514048);
  int*                hist     = (int*)(ws + 587776);
  int*                offs     = (int*)(ws + 604160);
  int*                sidx     = (int*)(ws + 620544);
  int*                keys     = (int*)(ws + 694272);
  float*              xin      = (float*)(ws + 768000);
  float*              xg       = (float*)(ws + 14923776);
  float*              xs_s     = (float*)(ws + 29079552);
  float*              delta_s  = (float*)(ws + 43235328);
  float*              Bs_s     = (float*)(ws + 57391104);
  float*              Cs_s     = (float*)(ws + 58570752);
  float*              y_s      = (float*)(ws + 59750400);

  hipMemsetAsync(argbuf, 0, (size_t)Bb * Nn * 8, stream);
  hipMemsetAsync(hist,   0, (size_t)Bb * NTOK * 4, stream);

  k_pack_bfrag<<<(128 * 512 + 255) / 256, 256, 0, stream>>>(rw2, frag_r2, 2048, 32, 1, 128 * 512);
  k_pack_bfrag<<<(36 * 512 + 255) / 256, 256, 0, stream>>>(ipw, frag_in, 192, 96, 3, 36 * 512);
  k_pack_bfrag<<<(18 * 512 + 255) / 256, 256, 0, stream>>>(xpw, frag_xp, 44, 192, 6, 18 * 512);
  k_pack_bfrag<<<(12 * 512 + 255) / 256, 256, 0, stream>>>(dtw, frag_dt, 192, 12, 1, 12 * 512);
  k_pack_bfrag<<<(36 * 512 + 255) / 256, 256, 0, stream>>>(ow, frag_out, 96, 192, 6, 36 * 512);

  k_full_emb<<<(NTOK * DST + 255) / 256, 256, 0, stream>>>(embB, tw, fe);
  k_route<<<dim3(Nn / 16, Bb), 256, 0, stream>>>(x, rw1, rb1, rb2, frag_r2, argbuf);
  k_extract<<<(Bb * Nn + 255) / 256, 256, 0, stream>>>(argbuf, idx, Bb * Nn);
  k_hist<<<(Bb * Nn + 255) / 256, 256, 0, stream>>>(idx, hist);
  k_offsets<<<Bb, 32, 0, stream>>>(hist, offs);
  k_place<<<dim3(Bb, 8), 256, 0, stream>>>(idx, offs, sidx, keys);

  k_inproj<<<dim3(Nn / 16, Bb), 384, 0, stream>>>(x, ipb, frag_in, xin);
  k_cpe<<<((size_t)Bb * Nn * HID + 255) / 256, 256, 0, stream>>>(xin, cpw, cpb, xg);
  k_xproj<<<dim3(Nn / 16, Bb), 384, 0, stream>>>(xg, sidx, keys, frag_xp, frag_dt, fe, dtb,
                                                 xs_s, delta_s, Bs_s, Cs_s);
  k_scan<<<dim3(HID / 16, Bb), 256, 0, stream>>>(Alog, Ds, xs_s, delta_s, Bs_s, Cs_s, y_s);
  k_out<<<dim3(Nn / 16, Bb), 256, 0, stream>>>(y_s, ng, nb, frag_out, ob, sidx, out);
}
